// PrefixAttentionWrapper_19370302505494
// MI455X (gfx1250) — matrix operand order repaired and checked by reading, not compiled, not run on hardware
//
#include <hip/hip_runtime.h>

typedef __bf16 bf16;
typedef __attribute__((ext_vector_type(8)))  __bf16 bf16x8;
typedef __attribute__((ext_vector_type(16))) __bf16 v16bf;
typedef __attribute__((ext_vector_type(8)))  float  v8f;
typedef __attribute__((ext_vector_type(4)))  float  f32x4;

#define B_   8
#define N_   1024
#define C_   768
#define H_   12
#define D_   64
#define P_   16
#define KV_  1040          // P_ + N_
#define KVP_ 1056          // padded to 33 * 32 key blocks
#define NBLK 33

// Build a 16-element bf16 fragment from two 8-element LDS/global chunks.
static __device__ inline v16bf frag16(const bf16* p0, const bf16* p1) {
  bf16x8 a = *(const bf16x8*)p0;
  bf16x8 b = *(const bf16x8*)p1;
  return __builtin_shufflevector(a, b, 0,1,2,3,4,5,6,7,8,9,10,11,12,13,14,15);
}

static __device__ inline v8f wmma_bf16(v16bf a, v16bf b, v8f c) {
  return __builtin_amdgcn_wmma_f32_16x16x32_bf16(false, a, false, b, (short)0, c,
                                                 false, false);
}

// Async global->LDS copy of 16 bytes (GVS mode: saddr + 32-bit voffset).
// Tracked by ASYNCcnt; drain with s_wait_asynccnt before the barrier.
static __device__ inline void async_copy_b128(unsigned lds_addr, unsigned goff,
                                              const bf16* sbase) {
  __asm__ volatile("global_load_async_to_lds_b128 %0, %1, %2"
                   :: "v"(lds_addr), "v"(goff), "s"(sbase) : "memory");
}

// ---------------------------------------------------------------------------
// Kernel 1: write bf16 prefix K/V rows (0..15) and zero pad rows (1040..1055)
// into concat K/V buffers laid out [B][H][KVP_][D].
// ---------------------------------------------------------------------------
__global__ void __launch_bounds__(256)
fill_prefix_kernel(const float* __restrict__ pk, const float* __restrict__ pv,
                   bf16* __restrict__ Kc, bf16* __restrict__ Vc) {
  int idx = blockIdx.x * 256 + threadIdx.x;   // < B_*H_*32*D_ = 196608
  int d  = idx & 63;
  int t  = idx >> 6;
  int r  = t & 31;
  int t2 = t >> 5;
  int h  = t2 % H_;
  int b  = t2 / H_;
  size_t base = ((size_t)(b * H_ + h)) * KVP_ * D_;
  if (r < P_) {
    int src = r * C_ + h * D_ + d;
    Kc[base + (size_t)r * D_ + d] = (bf16)pk[src];
    Vc[base + (size_t)r * D_ + d] = (bf16)pv[src];
  } else {
    int rr = KV_ + (r - P_);                  // zero the padding keys
    Kc[base + (size_t)rr * D_ + d] = (bf16)0.0f;
    Vc[base + (size_t)rr * D_ + d] = (bf16)0.0f;
  }
}

// ---------------------------------------------------------------------------
// Kernel 2: QKV GEMM.  out[m,f] = X[m,:] . W[f,:] + bias[f]
// m over B_*N_ (8192), f over 3*C_ (2304).  Scatter into Q / Kc / Vc (bf16).
// Block = 256 threads (8 waves); tile 256(M) x 64(F); K-step 32.
// Each wave owns 32 rows: 2 A-fragments reused against 4 B-fragments = 8 WMMA.
// ---------------------------------------------------------------------------
__global__ void __launch_bounds__(256)
qkv_gemm_kernel(const float* __restrict__ X, const float* __restrict__ W,
                const float* __restrict__ bias,
                bf16* __restrict__ Q, bf16* __restrict__ Kc, bf16* __restrict__ Vc) {
  __shared__ bf16 As[256 * 32];
  __shared__ bf16 Bs[64 * 32];

  int f0 = blockIdx.x * 64;
  int m0 = blockIdx.y * 256;
  int tid  = threadIdx.x;
  int wave = tid >> 5;
  int lane = tid & 31;
  int ln16 = lane & 15;
  bool hiH = lane >= 16;

  v8f acc[2][4] = {};

  for (int k0 = 0; k0 < C_; k0 += 32) {
    __syncthreads();
    {
      // A tile: each thread converts one full 32-wide row (fp32 -> bf16)
      const float* src = X + (size_t)(m0 + tid) * C_ + k0;
#pragma unroll
      for (int q = 0; q < 4; q++) {
        f32x4 a = *(const f32x4*)(src + q * 8);
        f32x4 b = *(const f32x4*)(src + q * 8 + 4);
        bf16x8 o;
#pragma unroll
        for (int j = 0; j < 4; j++) { o[j] = (bf16)a[j]; o[4 + j] = (bf16)b[j]; }
        *(bf16x8*)(As + tid * 32 + q * 8) = o;
      }
      // B tile: 64x32 weights fp32 -> bf16 (8 elements per thread)
      int brow = tid >> 2;
      int bcol = (tid & 3) * 8;
      const float* wsrc = W + (size_t)(f0 + brow) * C_ + k0 + bcol;
      f32x4 w0 = *(const f32x4*)(wsrc + 0);
      f32x4 w1 = *(const f32x4*)(wsrc + 4);
      bf16x8 wb;
#pragma unroll
      for (int j = 0; j < 4; j++) { wb[j] = (bf16)w0[j]; wb[4 + j] = (bf16)w1[j]; }
      *(bf16x8*)(Bs + brow * 32 + bcol) = wb;
    }
    __syncthreads();

    int akb = hiH ? 8 : 0;
    const bf16* ar0 = As + (wave * 32 + ln16) * 32;
    const bf16* ar1 = ar0 + 16 * 32;
    v16bf af0 = frag16(ar0 + akb, ar0 + akb + 16);
    v16bf af1 = frag16(ar1 + akb, ar1 + akb + 16);
#pragma unroll
    for (int g = 0; g < 4; g++) {
      const bf16* br = Bs + (g * 16 + ln16) * 32;
      int bkb = hiH ? 8 : 0;
      v16bf bfr = frag16(br + bkb, br + bkb + 16);
      acc[0][g] = wmma_bf16(af0, bfr, acc[0][g]);
      acc[1][g] = wmma_bf16(af1, bfr, acc[1][g]);
    }
  }

  // Epilogue: add bias, scatter to Q / concat-K / concat-V in bf16.
#pragma unroll
  for (int g = 0; g < 4; g++) {
    int f = f0 + g * 16 + ln16;
    float bia = bias[f];
    int which = f / C_;
    int rem   = f - which * C_;
    int h = rem >> 6;
    int d = rem & 63;
#pragma unroll
    for (int r = 0; r < 2; r++) {
#pragma unroll
      for (int i = 0; i < 8; i++) {
        int m = m0 + wave * 32 + r * 16 + (hiH ? 8 : 0) + i;
        int b = m >> 10;
        int n = m & 1023;
        bf16 o = (bf16)(acc[r][g][i] + bia);
        size_t bh = (size_t)(b * H_ + h);
        if (which == 0)      Q [ (bh * N_  + n      ) * D_ + d] = o;
        else if (which == 1) Kc[ (bh * KVP_ + P_ + n) * D_ + d] = o;
        else                 Vc[ (bh * KVP_ + P_ + n) * D_ + d] = o;
      }
    }
  }
}

// ---------------------------------------------------------------------------
// Kernel 3: flash attention.  Grid (N_/128, B_*H_), 256 threads (8 waves).
// Each wave: 16 query rows, online softmax over 33 key blocks of 32.
// K blocks are double-buffered via GLOBAL_LOAD_ASYNC_TO_LDS_B128 (ASYNCcnt);
// V blocks are double-buffered and staged transposed through registers.
// ---------------------------------------------------------------------------
__global__ void __launch_bounds__(256)
attn_kernel(const bf16* __restrict__ Q, const bf16* __restrict__ Kc,
            const bf16* __restrict__ Vc, bf16* __restrict__ Ob) {
  __shared__ bf16 Klds[2][32 * 64];   // [buf][key][d]   (async-filled)
  __shared__ bf16 Vt[2][64 * 32];     // [buf][d][key]   (transposed)
  __shared__ bf16 Pst[8][16 * 32];    // per-wave probability staging

  int bh = blockIdx.y;
  int b = bh / H_;
  int h = bh % H_;
  int q0 = blockIdx.x * 128;
  int tid  = threadIdx.x;
  int wave = tid >> 5;
  int lane = tid & 31;
  int ln16 = lane & 15;
  bool hiH = lane >= 16;

  const bf16* qbase = Q  + (size_t)(b * H_ + h) * N_  * D_;
  const bf16* kbase = Kc + (size_t)(b * H_ + h) * KVP_ * D_;
  const bf16* vbase = Vc + (size_t)(b * H_ + h) * KVP_ * D_;

  // LDS byte address of K buffers (flat shared addr low 32 bits == LDS addr)
  unsigned kldsbase = (unsigned)(size_t)(&Klds[0][0]);

  // Stage one 32-key block into buffer `buf`: K via async copy, V transposed.
  int skey = tid >> 3;               // 0..31
  int sd0  = (tid & 7) * 8;          // 0..56
  auto stage = [&](int blk, int buf) {
    unsigned loff = kldsbase + (unsigned)(buf * (32 * 64 * 2) + (skey * 64 + sd0) * 2);
    unsigned goff = (unsigned)(((blk * 32 + skey) * 64 + sd0) * 2);
    async_copy_b128(loff, goff, kbase);
    bf16x8 vv = *(const bf16x8*)(vbase + (size_t)(blk * 32 + skey) * D_ + sd0);
    bf16* vt = &Vt[buf][0];
#pragma unroll
    for (int j = 0; j < 8; j++) vt[(sd0 + j) * 32 + skey] = vv[j];
  };

  // Q fragments (A-layout) for this wave's 16 rows, D=64 -> two K=32 frags
  int kb0 = hiH ? 8 : 0;
  const bf16* qr = qbase + (size_t)(q0 + wave * 16 + ln16) * D_;
  v16bf qf0 = frag16(qr + kb0,      qr + kb0 + 16);
  v16bf qf1 = frag16(qr + 32 + kb0, qr + 32 + kb0 + 16);

  v8f acc[4] = {};
  float rmax[8], rsum[8];
#pragma unroll
  for (int i = 0; i < 8; i++) { rmax[i] = -1e30f; rsum[i] = 0.0f; }

  // Prologue: fill buffer 0 with block 0.
  stage(0, 0);
  __asm__ volatile("s_wait_asynccnt 0" ::: "memory");
  __syncthreads();

  int cur = 0;
  for (int blk = 0; blk < NBLK; ++blk) {
    int nxt = cur ^ 1;
    // Prefetch next block into the alternate buffer while we compute.
    if (blk + 1 < NBLK) stage(blk + 1, nxt);

    int krow0 = blk * 32;
    const bf16* kcur = &Klds[cur][0];
    const bf16* vcur = &Vt[cur][0];

    // S = Q K^T, two 16-key groups, contraction over D=64 (2 chained WMMA)
    v8f s[2];
#pragma unroll
    for (int g = 0; g < 2; g++) {
      const bf16* kr = kcur + (g * 16 + ln16) * 64;
      int db = hiH ? 8 : 0;
      v16bf kf0 = frag16(kr + db,      kr + db + 16);
      v16bf kf1 = frag16(kr + 32 + db, kr + 32 + db + 16);
      v8f cs = {};
      cs = wmma_bf16(qf0, kf0, cs);
      cs = wmma_bf16(qf1, kf1, cs);
      s[g] = cs;
    }

    // scale + tail mask (keys >= 1040 are padding)
#pragma unroll
    for (int g = 0; g < 2; g++) {
      bool masked = (krow0 + g * 16 + ln16) >= KV_;
#pragma unroll
      for (int i = 0; i < 8; i++) {
        float v = s[g][i] * 0.125f;    // D^-0.5
        s[g][i] = masked ? -1e30f : v;
      }
    }

    // online softmax: row stats live per 16-lane half (C-layout rows)
    float tmax[8];
#pragma unroll
    for (int i = 0; i < 8; i++) tmax[i] = fmaxf(s[0][i], s[1][i]);
#pragma unroll
    for (int off = 1; off < 16; off <<= 1)
#pragma unroll
      for (int i = 0; i < 8; i++)
        tmax[i] = fmaxf(tmax[i], __shfl_xor(tmax[i], off, 32));

    float alpha[8];
#pragma unroll
    for (int i = 0; i < 8; i++) {
      float nm = fmaxf(rmax[i], tmax[i]);
      alpha[i] = __expf(rmax[i] - nm);
      rmax[i]  = nm;
    }
#pragma unroll
    for (int g = 0; g < 2; g++)
#pragma unroll
      for (int i = 0; i < 8; i++) s[g][i] = __expf(s[g][i] - rmax[i]);

    float tsum[8];
#pragma unroll
    for (int i = 0; i < 8; i++) tsum[i] = s[0][i] + s[1][i];
#pragma unroll
    for (int off = 1; off < 16; off <<= 1)
#pragma unroll
      for (int i = 0; i < 8; i++) tsum[i] += __shfl_xor(tsum[i], off, 32);
#pragma unroll
    for (int i = 0; i < 8; i++) rsum[i] = rsum[i] * alpha[i] + tsum[i];
#pragma unroll
    for (int i = 0; i < 8; i++) {
      acc[0][i] *= alpha[i]; acc[1][i] *= alpha[i];
      acc[2][i] *= alpha[i]; acc[3][i] *= alpha[i];
    }

    // stage P (C-layout -> LDS row-major) then reload as A-fragment
    bf16* pw = Pst[wave];
    int prow = hiH ? 8 : 0;
#pragma unroll
    for (int g = 0; g < 2; g++)
#pragma unroll
      for (int i = 0; i < 8; i++)
        pw[(prow + i) * 32 + g * 16 + ln16] = (bf16)s[g][i];
    __asm__ volatile("s_wait_dscnt 0" ::: "memory");

    const bf16* pr = pw + ln16 * 32;
    int pkb = hiH ? 8 : 0;
    v16bf pf = frag16(pr + pkb, pr + pkb + 16);

    // O += P * V   (V as B-fragments from transposed LDS), 4 d-groups
#pragma unroll
    for (int dg = 0; dg < 4; dg++) {
      const bf16* vr = vcur + (dg * 16 + ln16) * 32;
      int vkb = hiH ? 8 : 0;
      v16bf vf = frag16(vr + vkb, vr + vkb + 16);
      acc[dg] = wmma_bf16(pf, vf, acc[dg]);
    }

    // Drain this wave's async prefetch, then rendezvous before buffer swap.
    __asm__ volatile("s_wait_asynccnt 0" ::: "memory");
    __syncthreads();
    cur = nxt;
  }

  // normalize and store bf16 output in [B][N][C] layout (c = h*64 + d)
  float inv[8];
#pragma unroll
  for (int i = 0; i < 8; i++) inv[i] = 1.0f / rsum[i];
#pragma unroll
  for (int dg = 0; dg < 4; dg++) {
    int d = dg * 16 + ln16;
#pragma unroll
    for (int i = 0; i < 8; i++) {
      int n = q0 + wave * 16 + (hiH ? 8 : 0) + i;
      Ob[((size_t)b * N_ + n) * C_ + h * D_ + d] = (bf16)(acc[dg][i] * inv[i]);
    }
  }
}

// ---------------------------------------------------------------------------
// Kernel 4: output projection.  out[m,f] = Ob[m,:] . Wp[f,:] + bias[f] (fp32)
// Tile 256(M) x 64(F); 8 WMMA per wave per K-step.
// ---------------------------------------------------------------------------
__global__ void __launch_bounds__(256)
proj_gemm_kernel(const bf16* __restrict__ A, const float* __restrict__ W,
                 const float* __restrict__ bias, float* __restrict__ Out) {
  __shared__ bf16 As[256 * 32];
  __shared__ bf16 Bs[64 * 32];

  int f0 = blockIdx.x * 64;
  int m0 = blockIdx.y * 256;
  int tid  = threadIdx.x;
  int wave = tid >> 5;
  int lane = tid & 31;
  int ln16 = lane & 15;
  bool hiH = lane >= 16;

  v8f acc[2][4] = {};

  for (int k0 = 0; k0 < C_; k0 += 32) {
    __syncthreads();
    {
      // A tile: one full bf16 row per thread
      const bf16* src = A + (size_t)(m0 + tid) * C_ + k0;
#pragma unroll
      for (int q = 0; q < 4; q++)
        *(bf16x8*)(As + tid * 32 + q * 8) = *(const bf16x8*)(src + q * 8);

      int brow = tid >> 2;
      int bcol = (tid & 3) * 8;
      const float* wsrc = W + (size_t)(f0 + brow) * C_ + k0 + bcol;
      f32x4 w0 = *(const f32x4*)(wsrc + 0);
      f32x4 w1 = *(const f32x4*)(wsrc + 4);
      bf16x8 wb;
#pragma unroll
      for (int j = 0; j < 4; j++) { wb[j] = (bf16)w0[j]; wb[4 + j] = (bf16)w1[j]; }
      *(bf16x8*)(Bs + brow * 32 + bcol) = wb;
    }
    __syncthreads();

    int akb = hiH ? 8 : 0;
    const bf16* ar0 = As + (wave * 32 + ln16) * 32;
    const bf16* ar1 = ar0 + 16 * 32;
    v16bf af0 = frag16(ar0 + akb, ar0 + akb + 16);
    v16bf af1 = frag16(ar1 + akb, ar1 + akb + 16);
#pragma unroll
    for (int g = 0; g < 4; g++) {
      const bf16* br = Bs + (g * 16 + ln16) * 32;
      int bkb = hiH ? 8 : 0;
      v16bf bfr = frag16(br + bkb, br + bkb + 16);
      acc[0][g] = wmma_bf16(af0, bfr, acc[0][g]);
      acc[1][g] = wmma_bf16(af1, bfr, acc[1][g]);
    }
  }

#pragma unroll
  for (int g = 0; g < 4; g++) {
    int f = f0 + g * 16 + ln16;
    float bia = bias[f];
#pragma unroll
    for (int r = 0; r < 2; r++) {
#pragma unroll
      for (int i = 0; i < 8; i++) {
        int m = m0 + wave * 32 + r * 16 + (hiH ? 8 : 0) + i;
        Out[(size_t)m * C_ + f] = acc[r][g][i] + bia;
      }
    }
  }
}

// ---------------------------------------------------------------------------
extern "C" void kernel_launch(void* const* d_in, const int* in_sizes, int n_in,
                              void* d_out, int out_size, void* d_ws, size_t ws_size,
                              hipStream_t stream) {
  const float* x        = (const float*)d_in[0];
  const float* qkv_w    = (const float*)d_in[1];
  const float* qkv_b    = (const float*)d_in[2];
  const float* proj_w   = (const float*)d_in[3];
  const float* proj_b   = (const float*)d_in[4];
  const float* prefix_k = (const float*)d_in[5];
  const float* prefix_v = (const float*)d_in[6];
  float* out = (float*)d_out;

  // workspace layout (bf16): Q | Kc | Vc | Ob   (~51 MB total)
  bf16* Q  = (bf16*)d_ws;
  bf16* Kc = Q  + (size_t)B_ * H_ * N_   * D_;   // 6291456 elems
  bf16* Vc = Kc + (size_t)B_ * H_ * KVP_ * D_;   // 6488064 elems
  bf16* Ob = Vc + (size_t)B_ * H_ * KVP_ * D_;   // 6488064 elems

  fill_prefix_kernel<<<dim3((B_ * H_ * 32 * D_) / 256), dim3(256), 0, stream>>>(
      prefix_k, prefix_v, Kc, Vc);

  qkv_gemm_kernel<<<dim3((3 * C_) / 64, (B_ * N_) / 256), dim3(256), 0, stream>>>(
      x, qkv_w, qkv_b, Q, Kc, Vc);

  attn_kernel<<<dim3(N_ / 128, B_ * H_), dim3(256), 0, stream>>>(Q, Kc, Vc, Ob);

  proj_gemm_kernel<<<dim3(C_ / 64, (B_ * N_) / 256), dim3(256), 0, stream>>>(
      Ob, proj_w, proj_b, out);
}
